// _fasterRCNNAttentionMinus_58909771431947
// MI455X (gfx1250) — matrix operand order repaired and checked by reading, not compiled, not run on hardware
//
#include <hip/hip_runtime.h>

typedef __bf16 bf16;
typedef __attribute__((ext_vector_type(16))) __bf16 v16bf;
typedef __attribute__((ext_vector_type(8)))  __bf16 v8bf;
typedef __attribute__((ext_vector_type(8)))  float  v8f;

#define NTOK  4096
#define FEAT  1024
#define KIN   2048
#define NONGT 300
#define NPAD  304
#define GROUP 16
#define DG    64

// ---------------- elementwise f32 -> bf16 ----------------
__global__ void k_cvt_bf16(const float* __restrict__ src, bf16* __restrict__ dst, long n) {
  long i = (long)blockIdx.x * blockDim.x + threadIdx.x;
  long stride = (long)gridDim.x * blockDim.x;
  for (; i < n; i += stride) dst[i] = (bf16)src[i];
}

// nongt_feat = x[:300] padded to 304 rows (zeros), bf16
__global__ void k_nongt(const float* __restrict__ x, bf16* __restrict__ dst) {
  int i = blockIdx.x * blockDim.x + threadIdx.x;
  if (i >= NPAD * FEAT) return;
  int r = i / FEAT;
  dst[i] = (r < NONGT) ? (bf16)x[i] : (bf16)0.0f;
}

// kT[g][d][m] = kproj[m][g*64+d], zero-padded m>=300
__global__ void k_kT(const float* __restrict__ kproj, bf16* __restrict__ kT) {
  int i = blockIdx.x * blockDim.x + threadIdx.x;
  if (i >= GROUP * DG * NPAD) return;
  int g = i / (DG * NPAD);
  int rem = i % (DG * NPAD);
  int d = rem / NPAD;
  int m = rem % NPAD;
  kT[i] = (m < NONGT) ? (bf16)kproj[(long)m * FEAT + g * DG + d] : (bf16)0.0f;
}

// wcT[g][c][o] = wc[g][o][c]  (wc: (16,64,1024))
__global__ void k_wcT(const float* __restrict__ wc, bf16* __restrict__ wcT) {
  int i = blockIdx.x * blockDim.x + threadIdx.x;
  if (i >= GROUP * FEAT * DG) return;
  int g = i / (FEAT * DG);
  int rem = i % (FEAT * DG);
  int c = rem / DG;
  int o = rem % DG;
  wcT[i] = (bf16)wc[((long)g * DG + o) * FEAT + c];
}

// W[g][n][m] = log(max(relu(pe[n,m,:] . wp[:,g] + bp[g]), 1e-6))
__global__ __launch_bounds__(128)
void k_pos_logits(const float* __restrict__ pe, const float* __restrict__ wp,
                  const float* __restrict__ bp, float* __restrict__ W) {
  __shared__ float rowbuf[8][64];
  __shared__ float wps[64][16];
  for (int i = threadIdx.x; i < 64 * 16; i += 128) wps[i >> 4][i & 15] = wp[i];
  long blockRow = (long)blockIdx.x * 8;
  const long totalRows = (long)NTOK * NONGT;
  for (int i = threadIdx.x; i < 8 * 64; i += 128) {
    long r = blockRow + (i >> 6);
    rowbuf[i >> 6][i & 63] = (r < totalRows) ? pe[r * 64 + (i & 63)] : 0.0f;
  }
  __syncthreads();
  int g = threadIdx.x & 15;
  int lr = threadIdx.x >> 4;
  long r = blockRow + lr;
  if (r < totalRows) {
    float acc = bp[g];
#pragma unroll
    for (int d = 0; d < 64; ++d) acc = fmaf(rowbuf[lr][d], wps[d][g], acc);
    float v = fmaxf(acc, 0.0f);
    v = __logf(fmaxf(v, 1e-6f));
    int n = (int)(r / NONGT), m = (int)(r % NONGT);
    W[((long)g * NTOK + n) * NPAD + m] = v;
  }
}

// wave-per-row softmax over m<300; S[(n*16+g)][m] bf16, pad -> 0
__global__ __launch_bounds__(256)
void k_softmax(const float* __restrict__ W, bf16* __restrict__ S) {
  int row = blockIdx.x * 8 + (threadIdx.x >> 5);
  int lane = threadIdx.x & 31;
  int n = row >> 4;
  int g = row & 15;
  const float* w = W + ((long)g * NTOK + n) * NPAD;
  float vals[10];
  float mx = -3.0e38f;
  int cnt = 0;
  for (int m = lane; m < NONGT; m += 32) {
    float v = w[m];
    vals[cnt++] = v;
    mx = fmaxf(mx, v);
  }
#pragma unroll
  for (int off = 16; off >= 1; off >>= 1) mx = fmaxf(mx, __shfl_xor(mx, off));
  float sum = 0.0f;
  for (int i = 0; i < cnt; ++i) { vals[i] = __expf(vals[i] - mx); sum += vals[i]; }
#pragma unroll
  for (int off = 16; off >= 1; off >>= 1) sum += __shfl_xor(sum, off);
  float inv = 1.0f / sum;
  bf16* s = S + (long)row * NPAD;
  cnt = 0;
  for (int m = lane; m < NPAD; m += 32) {
    float v = (m < NONGT) ? vals[cnt++] * inv : 0.0f;
    s[m] = (bf16)v;
  }
}

// ---------------- WMMA bf16 GEMM:  D = alpha*(A@B) + beta*Cin + bscale*bias ----------------
// Block tile 64x64, 128 threads = 4 waves, each wave 32x32 via 2x2 v_wmma_f32_16x16x32_bf16.
// A staged row-major (m,k); B staged TRANSPOSED (n,k) so both fragment gathers are
// contiguous 16-byte LDS reads (ds_load_b128). Full tiles use uint4 (8xbf16) global loads.
template <bool OUT_BF16>
__global__ __launch_bounds__(128)
void k_gemm(const bf16* __restrict__ A, int lda,
            const bf16* __restrict__ B, int ldb,
            void* __restrict__ Dv, int ldd,
            const float* __restrict__ Cin, int ldc,
            const float* __restrict__ bias,
            float alpha, float beta, float bscale,
            int M, int N, int K) {
  __shared__ __align__(16) bf16 As[64][40];   // (m, k), row = 80B (16B aligned, 20-bank stride)
  __shared__ __align__(16) bf16 BsT[64][40];  // (n, k) transposed
  int tid = threadIdx.x;
  int lane = tid & 31;
  int wave = tid >> 5;
  int wm = (wave >> 1) * 32;
  int wn = (wave & 1) * 32;
  int bm = blockIdx.y * 64;
  int bn = blockIdx.x * 64;
  int lm = lane & 15;
  int lh = lane >> 4;

  const bool fullM = (bm + 64 <= M);
  const bool fullN = (bn + 64 <= N);

  v8f acc[2][2] = {};
  for (int k0 = 0; k0 < K; k0 += 32) {
    const bool fullK = (k0 + 32 <= K);
    // ---- stage A tile (64 x 32) ----
    if (fullM && fullK) {
#pragma unroll
      for (int i = 0; i < 2; ++i) {
        int linear = tid + i * 128;              // 0..255
        int r = linear >> 2, c = (linear & 3) * 8;
        *(uint4*)&As[r][c] = *(const uint4*)&A[(long)(bm + r) * lda + k0 + c];
      }
    } else {
      for (int idx = tid; idx < 64 * 32; idx += 128) {
        int r = idx >> 5, c = idx & 31;
        int gr = bm + r, gc = k0 + c;
        bf16 val = A[(long)(gr < M ? gr : M - 1) * lda + (gc < K ? gc : K - 1)];
        As[r][c] = (gr < M && gc < K) ? val : (bf16)0.0f;
      }
    }
    // ---- stage B tile (32 x 64) transposed into BsT ----
    if (fullN && fullK) {
#pragma unroll
      for (int i = 0; i < 2; ++i) {
        int linear = tid + i * 128;              // 0..255
        int r = linear >> 3, c = (linear & 7) * 8;   // r = k (0..31), c = n base
        uint4 v = *(const uint4*)&B[(long)(k0 + r) * ldb + bn + c];
        bf16 tmp[8];
        *(uint4*)tmp = v;
#pragma unroll
        for (int e = 0; e < 8; ++e) BsT[c + e][r] = tmp[e];
      }
    } else {
      for (int idx = tid; idx < 32 * 64; idx += 128) {
        int r = idx >> 6, c = idx & 63;
        int gr = k0 + r, gc = bn + c;
        bf16 val = B[(long)(gr < K ? gr : K - 1) * ldb + (gc < N ? gc : N - 1)];
        BsT[c][r] = (gr < K && gc < N) ? val : (bf16)0.0f;
      }
    }
    __syncthreads();
    if (k0 + 32 < K) {  // prefetch next K tile -> global_prefetch_b8
      int pr = bm + (tid & 63);
      if (pr < M) __builtin_prefetch(&A[(long)pr * lda + k0 + 32], 0, 1);
      int br = k0 + 32 + (tid & 31);
      if (br < K && bn < N) __builtin_prefetch(&B[(long)br * ldb + bn], 0, 1);
    }
    // ---- fragments: each is two contiguous 16B LDS reads ----
    // A lane layout: row = wm(+16)+lm, K = lh*8 + {0..7} and 16 + lh*8 + {0..7}
    v8bf a0lo = *(const v8bf*)&As[wm + lm][lh * 8];
    v8bf a0hi = *(const v8bf*)&As[wm + lm][lh * 8 + 16];
    v8bf a1lo = *(const v8bf*)&As[wm + 16 + lm][lh * 8];
    v8bf a1hi = *(const v8bf*)&As[wm + 16 + lm][lh * 8 + 16];
    v16bf a0 = __builtin_shufflevector(a0lo, a0hi, 0, 1, 2, 3, 4, 5, 6, 7, 8, 9, 10, 11, 12, 13, 14, 15);
    v16bf a1 = __builtin_shufflevector(a1lo, a1hi, 0, 1, 2, 3, 4, 5, 6, 7, 8, 9, 10, 11, 12, 13, 14, 15);
    // B lane layout: col = wn(+16)+lm, K = lh*8 + {0..7} and 16 + lh*8 + {0..7}
    v8bf b0lo = *(const v8bf*)&BsT[wn + lm][lh * 8];
    v8bf b0hi = *(const v8bf*)&BsT[wn + lm][lh * 8 + 16];
    v8bf b1lo = *(const v8bf*)&BsT[wn + 16 + lm][lh * 8];
    v8bf b1hi = *(const v8bf*)&BsT[wn + 16 + lm][lh * 8 + 16];
    v16bf b0 = __builtin_shufflevector(b0lo, b0hi, 0, 1, 2, 3, 4, 5, 6, 7, 8, 9, 10, 11, 12, 13, 14, 15);
    v16bf b1 = __builtin_shufflevector(b1lo, b1hi, 0, 1, 2, 3, 4, 5, 6, 7, 8, 9, 10, 11, 12, 13, 14, 15);

    acc[0][0] = __builtin_amdgcn_wmma_f32_16x16x32_bf16(false, a0, false, b0, (short)0, acc[0][0], false, false);
    acc[0][1] = __builtin_amdgcn_wmma_f32_16x16x32_bf16(false, a0, false, b1, (short)0, acc[0][1], false, false);
    acc[1][0] = __builtin_amdgcn_wmma_f32_16x16x32_bf16(false, a1, false, b0, (short)0, acc[1][0], false, false);
    acc[1][1] = __builtin_amdgcn_wmma_f32_16x16x32_bf16(false, a1, false, b1, (short)0, acc[1][1], false, false);
    __syncthreads();
  }
  // C/D layout: VGPR e -> M = e + 8*(lane/16), N = lane%16
#pragma unroll
  for (int i = 0; i < 2; ++i)
#pragma unroll
    for (int j = 0; j < 2; ++j)
#pragma unroll
      for (int e = 0; e < 8; ++e) {
        int row = bm + wm + i * 16 + e + (lh << 3);
        int col = bn + wn + j * 16 + lm;
        if (row < M && col < N) {
          float val = alpha * acc[i][j][e];
          if (Cin)  val += beta * Cin[(long)row * ldc + col];
          if (bias) val += bscale * bias[col];
          if (OUT_BF16) ((bf16*)Dv)[(long)row * ldd + col] = (bf16)val;
          else          ((float*)Dv)[(long)row * ldd + col] = val;
        }
      }
}

extern "C" void kernel_launch(void* const* d_in, const int* in_sizes, int n_in,
                              void* d_out, int out_size, void* d_ws, size_t ws_size,
                              hipStream_t stream) {
  (void)in_sizes; (void)n_in; (void)out_size; (void)ws_size;
  const float* pooled = (const float*)d_in[0];
  const float* pe     = (const float*)d_in[1];
  const float* fc1_w  = (const float*)d_in[2];
  const float* fc1_b  = (const float*)d_in[3];
  const float* fc2_w  = (const float*)d_in[4];
  const float* fc2_b  = (const float*)d_in[5];
  const float* att_wp = (const float*)d_in[6];
  const float* att_bp = (const float*)d_in[7];
  const float* att_wq = (const float*)d_in[8];
  const float* att_bq = (const float*)d_in[9];
  const float* att_wk = (const float*)d_in[10];
  const float* att_bk = (const float*)d_in[11];
  const float* att_wc = (const float*)d_in[12];
  const float* att_bc = (const float*)d_in[13];
  float* out = (float*)d_out;

  char* ws = (char*)d_ws;
  size_t off = 0;
  auto alloc = [&](size_t bytes) -> char* {
    off = (off + 255) & ~(size_t)255;
    char* p = ws + off;
    off += bytes;
    return p;
  };
  bf16*  xbf2048 = (bf16*)alloc((size_t)NTOK * KIN * 2);
  bf16*  wfc1    = (bf16*)alloc((size_t)KIN * FEAT * 2);
  bf16*  wfc2    = (bf16*)alloc((size_t)FEAT * FEAT * 2);
  bf16*  wq      = (bf16*)alloc((size_t)4 * FEAT * FEAT * 2);
  bf16*  wk      = (bf16*)alloc((size_t)4 * FEAT * FEAT * 2);
  float* P1      = (float*)alloc((size_t)NTOK * FEAT * 4);
  float* P2      = (float*)alloc((size_t)NTOK * FEAT * 4);
  float* P3      = (float*)alloc((size_t)NTOK * FEAT * 4);
  bf16*  pbf     = (bf16*)alloc((size_t)NTOK * FEAT * 2);
  bf16*  qbf     = (bf16*)alloc((size_t)NTOK * FEAT * 2);
  float* kproj   = (float*)alloc((size_t)NPAD * FEAT * 4);
  bf16*  kT      = (bf16*)alloc((size_t)GROUP * DG * NPAD * 2);
  bf16*  nongt   = (bf16*)alloc((size_t)NPAD * FEAT * 2);
  bf16*  wcT     = (bf16*)alloc((size_t)GROUP * FEAT * DG * 2);
  float* W       = (float*)alloc((size_t)GROUP * NTOK * NPAD * 4);
  bf16*  S       = (bf16*)alloc((size_t)NTOK * GROUP * NPAD * 2);
  bf16*  Tg      = (bf16*)alloc((size_t)NTOK * FEAT * 2);

  auto cvt = [&](const float* s, bf16* d, long n) {
    int blocks = (int)((n + 255) / 256);
    k_cvt_bf16<<<blocks, 256, 0, stream>>>(s, d, n);
  };
  auto gemmF = [&](const bf16* A, int lda, const bf16* B, int ldb, float* D, int ldd,
                   const float* Cin, int ldc, const float* bias,
                   float alpha, float beta, float bscale, int M, int N, int K) {
    dim3 grid((N + 63) / 64, (M + 63) / 64);
    k_gemm<false><<<grid, 128, 0, stream>>>(A, lda, B, ldb, (void*)D, ldd, Cin, ldc, bias,
                                            alpha, beta, bscale, M, N, K);
  };
  auto gemmB = [&](const bf16* A, int lda, const bf16* B, int ldb, bf16* D, int ldd,
                   const float* Cin, int ldc, const float* bias,
                   float alpha, float beta, float bscale, int M, int N, int K) {
    dim3 grid((N + 63) / 64, (M + 63) / 64);
    k_gemm<true><<<grid, 128, 0, stream>>>(A, lda, B, ldb, (void*)D, ldd, Cin, ldc, bias,
                                           alpha, beta, bscale, M, N, K);
  };

  // attention block ai: dst[:, g*64:...] = Cin0 + sign*(attn + bc)
  auto run_attention = [&](int ai, const float* Cin0, float* dst, float sign) {
    k_pos_logits<<<(NTOK * NONGT + 7) / 8, 128, 0, stream>>>(
        pe, att_wp + (size_t)ai * 64 * GROUP, att_bp + ai * GROUP, W);
    gemmB(pbf, FEAT, wq + (size_t)ai * FEAT * FEAT, FEAT, qbf, FEAT,
          nullptr, 0, att_bq + ai * FEAT, 1.f, 0.f, 1.f, NTOK, FEAT, FEAT);
    gemmF(pbf, FEAT, wk + (size_t)ai * FEAT * FEAT, FEAT, kproj, FEAT,
          nullptr, 0, att_bk + ai * FEAT, 1.f, 0.f, 1.f, NPAD, FEAT, FEAT);
    k_kT<<<(GROUP * DG * NPAD + 255) / 256, 256, 0, stream>>>(kproj, kT);
    for (int g = 0; g < GROUP; ++g)
      gemmF(qbf + g * DG, FEAT, kT + (size_t)g * DG * NPAD, NPAD,
            W + (size_t)g * NTOK * NPAD, NPAD,
            W + (size_t)g * NTOK * NPAD, NPAD, nullptr,
            0.125f, 1.f, 0.f, NTOK, NPAD, DG);
    k_softmax<<<(NTOK * GROUP) / 8, 256, 0, stream>>>(W, S);
    k_wcT<<<(GROUP * FEAT * DG + 255) / 256, 256, 0, stream>>>(
        att_wc + (size_t)ai * GROUP * DG * FEAT, wcT);
    for (int g = 0; g < GROUP; ++g) {
      gemmB(S + g * NPAD, GROUP * NPAD, nongt, FEAT, Tg, FEAT,
            nullptr, 0, nullptr, 1.f, 0.f, 0.f, NTOK, FEAT, NPAD);
      gemmF(Tg, FEAT, wcT + (size_t)g * FEAT * DG, DG, dst + g * DG, FEAT,
            Cin0 + g * DG, FEAT, att_bc + ai * FEAT + g * DG,
            sign, 1.f, sign, NTOK, DG, FEAT);
    }
  };

  // dst = x + attn_a0(x) - attn_a1(x)   (both attentions use the SAME input x)
  auto run_pair = [&](const float* x, float* dst, int a0, int a1) {
    cvt(x, pbf, (long)NTOK * FEAT);
    k_nongt<<<(NPAD * FEAT + 255) / 256, 256, 0, stream>>>(x, nongt);
    run_attention(a0, x, dst, +1.f);
    run_attention(a1, dst, dst, -1.f);
  };

  // weight / input conversions
  cvt(pooled, xbf2048, (long)NTOK * KIN);
  cvt(fc1_w, wfc1, (long)KIN * FEAT);
  cvt(fc2_w, wfc2, (long)FEAT * FEAT);
  cvt(att_wq, wq, (long)4 * FEAT * FEAT);
  cvt(att_wk, wk, (long)4 * FEAT * FEAT);

  // fc1
  gemmF(xbf2048, KIN, wfc1, FEAT, P1, FEAT, nullptr, 0, fc1_b, 1.f, 0.f, 1.f, NTOK, FEAT, KIN);
  // pair 1 (attn 0, 1)
  run_pair(P1, P2, 0, 1);
  // fc2
  cvt(P2, pbf, (long)NTOK * FEAT);
  gemmF(pbf, FEAT, wfc2, FEAT, P3, FEAT, nullptr, 0, fc2_b, 1.f, 0.f, 1.f, NTOK, FEAT, FEAT);
  // pair 2 (attn 2, 3) -> final output
  run_pair(P3, out, 2, 3);
}
